// NoveltyBuffer_23587960389770
// MI455X (gfx1250) — compile-verified, matrix operand order checked by reading, not run on hardware
//
#include <hip/hip_runtime.h>
#include <cstdint>
#include <cstddef>

// ---------------------------------------------------------------------------
// MI455X (gfx1250, wave32) NoveltyBuffer kernel.
//   sx  = rownorm(x @ R)            [2048 x 256]
//   out = 1 - max_n (sx @ buffer^T) [2048]
// bf16x3 split GEMMs on v_wmma_f32_16x16x32_bf16; buffer streamed once from
// HBM (fits the 192MB L2), register-resident B fragments, LDS-staged A
// fragments, running-max reduction instead of materializing sim.
// v2: independent WMMA accumulation chains (HH / HL+LH, 2 M-tiles per
// iteration) so the XDL pipe has ILP instead of v_nop hazard stalls; half the
// barrier count.
// ---------------------------------------------------------------------------

typedef __attribute__((ext_vector_type(16))) __bf16 v16bf;
typedef __attribute__((ext_vector_type(8)))  __bf16 v8bf;
typedef __attribute__((ext_vector_type(8)))  float  v8f;

#define WMMA_BF16(a, b, c) \
  __builtin_amdgcn_wmma_f32_16x16x32_bf16(false, (a), false, (b), (short)0, (c), false, false)

constexpr int B_ROWS = 2048;    // batch rows
constexpr int DIM    = 2048;    // input dim (K of first GEMM)
constexpr int SK     = 256;     // sketch dim (K of second GEMM)
constexpr int NBUF   = 131072;  // buffer rows (N of second GEMM)

// Order-preserving float <-> uint encoding so we can use integer atomicMax.
__device__ __forceinline__ unsigned encode_f32(float f) {
  unsigned b = __float_as_uint(f);
  return (b & 0x80000000u) ? ~b : (b | 0x80000000u);
}
__device__ __forceinline__ float decode_f32(unsigned u) {
  unsigned b = (u & 0x80000000u) ? (u ^ 0x80000000u) : ~u;
  return __uint_as_float(b);
}

// Split f32 into bf16 hi + bf16 lo residual (RNE).
__device__ __forceinline__ void split2(float f, __bf16& h, __bf16& l) {
  h = (__bf16)f;
  l = (__bf16)(f - (float)h);
}

// ---------------------------------------------------------------------------
// Kernel 0: init global max keys to encode(-2.0)  (sim >= -1 always)
// ---------------------------------------------------------------------------
__global__ void k_init(unsigned* __restrict__ gkey) {
  int i = blockIdx.x * 256 + threadIdx.x;
  if (i < B_ROWS) gkey[i] = encode_f32(-2.0f);
}

// ---------------------------------------------------------------------------
// Kernel 1: transpose R [DIM x SK] -> Rt hi/lo bf16 [SK x DIM]
// (makes B-matrix fragment loads in the sketch GEMM contiguous along K)
// ---------------------------------------------------------------------------
__global__ void k_transposeR(const float* __restrict__ R,
                             __bf16* __restrict__ Rth,
                             __bf16* __restrict__ Rtl) {
  int idx = blockIdx.x * 256 + threadIdx.x;   // DIM*SK threads
  int k = idx >> 8;         // SK == 256
  int n = idx & 255;
  float v = R[idx];
  __bf16 h, l;
  split2(v, h, l);
  Rth[(size_t)n * DIM + k] = h;
  Rtl[(size_t)n * DIM + k] = l;
}

// ---------------------------------------------------------------------------
// Kernel 2: sketch GEMM sx = x @ R (bf16x3 WMMA), row L2-normalize,
// write sx as bf16 hi/lo [B_ROWS x SK].
// One workgroup = 16 rows of x; 8 waves each own two 16-col tiles of SK.
// Two independent accumulation chains per tile (HH, HL+LH) -> 4 chains/wave.
// ---------------------------------------------------------------------------
__global__ void __launch_bounds__(256) k_sketch(const float*  __restrict__ x,
                                                const __bf16* __restrict__ Rth,
                                                const __bf16* __restrict__ Rtl,
                                                __bf16* __restrict__ sxh,
                                                __bf16* __restrict__ sxl) {
  const int tid  = threadIdx.x;
  const int w    = tid >> 5;        // wave id (0..7), warpSize == 32
  const int l    = tid & 31;        // lane
  const int lrow = l & 15;
  const bool hih = (l >= 16);
  // 16-bit A/B fragment K-run offsets per lane half (per ISA 16-bit layout)
  const int off0 = hih ? 8 : 0;
  const int off1 = hih ? 24 : 16;
  const int mbase = blockIdx.x * 16;

  const v8f vzero = (v8f){0.f,0.f,0.f,0.f,0.f,0.f,0.f,0.f};
  v8f accHH[2] = {vzero, vzero};   // hi*hi chain
  v8f accML[2] = {vzero, vzero};   // hi*lo + lo*hi chain

  const float* xrow = x + (size_t)(mbase + lrow) * DIM;

  for (int k0 = 0; k0 < DIM; k0 += 32) {
    // A fragment: 16x32 f32 tile of x, converted to bf16 hi/lo on the fly
    v16bf ah, al;
#pragma unroll
    for (int j = 0; j < 8; ++j) {
      float v0 = xrow[k0 + off0 + j];
      float v1 = xrow[k0 + off1 + j];
      __bf16 h, lo2;
      split2(v0, h, lo2); ah[j]     = h; al[j]     = lo2;
      split2(v1, h, lo2); ah[j + 8] = h; al[j + 8] = lo2;
    }
#pragma unroll
    for (int t = 0; t < 2; ++t) {
      const int col = (2 * w + t) * 16 + lrow;
      const __bf16* bh = Rth + (size_t)col * DIM + k0;
      const __bf16* bl = Rtl + (size_t)col * DIM + k0;
      v8bf h0 = *(const v8bf*)(bh + off0);
      v8bf h1 = *(const v8bf*)(bh + off1);
      v8bf l0 = *(const v8bf*)(bl + off0);
      v8bf l1 = *(const v8bf*)(bl + off1);
      v16bf Bh = __builtin_shufflevector(h0, h1, 0,1,2,3,4,5,6,7,8,9,10,11,12,13,14,15);
      v16bf Bl = __builtin_shufflevector(l0, l1, 0,1,2,3,4,5,6,7,8,9,10,11,12,13,14,15);
      accHH[t] = WMMA_BF16(ah, Bh, accHH[t]);   // hi*hi
      accML[t] = WMMA_BF16(ah, Bl, accML[t]);   // hi*lo
      accML[t] = WMMA_BF16(al, Bh, accML[t]);   // lo*hi
    }
  }

  v8f acc[2];
#pragma unroll
  for (int t = 0; t < 2; ++t)
#pragma unroll
    for (int i = 0; i < 8; ++i) acc[t][i] = accHH[t][i] + accML[t][i];

  // Row sums of squares across all 256 columns (shared by the 8 waves).
  __shared__ float rowsq[16];
  if (tid < 16) rowsq[tid] = 0.0f;
  __syncthreads();
#pragma unroll
  for (int t = 0; t < 2; ++t)
#pragma unroll
    for (int i = 0; i < 8; ++i) {
      float v = acc[t][i];
      atomicAdd(&rowsq[i + (hih ? 8 : 0)], v * v);   // ds_add_f32
    }
  __syncthreads();

  // Normalize (matches x / max(||x||, 1e-12)), split to hi/lo, store.
#pragma unroll
  for (int t = 0; t < 2; ++t) {
    const int col = (2 * w + t) * 16 + lrow;
#pragma unroll
    for (int i = 0; i < 8; ++i) {
      const int r = i + (hih ? 8 : 0);
      float s = 1.0f / fmaxf(sqrtf(rowsq[r]), 1e-12f);
      float v = acc[t][i] * s;
      __bf16 h, lo2;
      split2(v, h, lo2);
      size_t o = (size_t)(mbase + r) * SK + col;
      sxh[o] = h;
      sxl[o] = lo2;
    }
  }
}

// ---------------------------------------------------------------------------
// Kernel 3: max-similarity GEMM. Workgroup owns 128 buffer rows (N), loops
// over all 2048 sx rows (M), TWO 16-row M-tiles per iteration. buffer read
// ONCE from HBM, held in registers as bf16 hi/lo WMMA fragments (128 VGPRs).
// sx staged per iteration into LDS in fragment order. Four independent WMMA
// accumulation chains per wave (2 tiles x {HH, HL+LH}) keep the XDL pipe fed.
// Per-row max reduced via shuffles + LDS uint atomics, merged globally.
// ---------------------------------------------------------------------------
__global__ void __launch_bounds__(256) k_simmax(const __bf16* __restrict__ sxh,
                                                const __bf16* __restrict__ sxl,
                                                const float*  __restrict__ buf,
                                                unsigned* __restrict__ gkey) {
  const int tid  = threadIdx.x;
  const int w    = tid >> 5;
  const int l    = tid & 31;
  const int lrow = l & 15;
  const bool hih = (l >= 16);
  const int off0 = hih ? 8 : 0;
  const int off1 = hih ? 24 : 16;
  const int nb   = blockIdx.x * 128;          // this WG's buffer-row base

  __shared__ unsigned lmax[B_ROWS];                         // 8 KB
  __shared__ alignas(32) __bf16 aHi[2][8][32][16];          // 16 KB
  __shared__ alignas(32) __bf16 aLo[2][8][32][16];          // 16 KB

  // --- Preload this wave's 16 buffer rows as register-resident B fragments.
  v16bf bhi[8], blo[8];
  {
    const float* brow = buf + (size_t)(nb + w * 16 + lrow) * SK;
#pragma unroll
    for (int kc = 0; kc < 8; ++kc) {
#pragma unroll
      for (int j = 0; j < 8; ++j) {
        float v0 = brow[kc * 32 + off0 + j];
        float v1 = brow[kc * 32 + off1 + j];
        __bf16 h, lo2;
        split2(v0, h, lo2); bhi[kc][j]     = h; blo[kc][j]     = lo2;
        split2(v1, h, lo2); bhi[kc][j + 8] = h; blo[kc][j + 8] = lo2;
      }
    }
  }

  const unsigned NEG = encode_f32(-2.0f);
#pragma unroll
  for (int i = 0; i < 8; ++i) lmax[tid * 8 + i] = NEG;
  __syncthreads();

  const v8f vzero = (v8f){0.f,0.f,0.f,0.f,0.f,0.f,0.f,0.f};

  for (int mt2 = 0; mt2 < B_ROWS / 32; ++mt2) {
    // Cooperative stage of TWO A tiles (32 rows x 256 K, hi/lo) in fragment
    // order: 256 threads each fill one (kchunk, lane) slot per tile.
    {
      const int kc = w, ll = l;
      const int o0 = (ll >= 16) ? 8 : 0, o1 = (ll >= 16) ? 24 : 16;
#pragma unroll
      for (int t = 0; t < 2; ++t) {
        const int srow = mt2 * 32 + t * 16 + (ll & 15);
        const __bf16* sh = sxh + (size_t)srow * SK + kc * 32;
        const __bf16* sl = sxl + (size_t)srow * SK + kc * 32;
        *(uint4*)&aHi[t][kc][ll][0] = *(const uint4*)(sh + o0);
        *(uint4*)&aHi[t][kc][ll][8] = *(const uint4*)(sh + o1);
        *(uint4*)&aLo[t][kc][ll][0] = *(const uint4*)(sl + o0);
        *(uint4*)&aLo[t][kc][ll][8] = *(const uint4*)(sl + o1);
      }
      if (mt2 + 1 < B_ROWS / 32) {  // pull next iteration's sx toward L0/L2
        __builtin_prefetch(sxh + (size_t)((mt2 + 1) * 32 + (ll & 15)) * SK + kc * 32, 0, 1);
      }
    }
    __syncthreads();

    v8f accHH[2] = {vzero, vzero};
    v8f accML[2] = {vzero, vzero};
#pragma unroll
    for (int kc = 0; kc < 8; ++kc) {
#pragma unroll
      for (int t = 0; t < 2; ++t) {
        v16bf ah = *(const v16bf*)&aHi[t][kc][l][0];
        v16bf al = *(const v16bf*)&aLo[t][kc][l][0];
        accHH[t] = WMMA_BF16(ah, bhi[kc], accHH[t]);   // hi*hi
        accML[t] = WMMA_BF16(ah, blo[kc], accML[t]);   // hi*lo
        accML[t] = WMMA_BF16(al, bhi[kc], accML[t]);   // lo*hi
      }
    }

    // Max over this wave's 16 N-columns for each of the 32 M rows.
#pragma unroll
    for (int t = 0; t < 2; ++t) {
#pragma unroll
      for (int i = 0; i < 8; ++i) {
        float v = accHH[t][i] + accML[t][i];
#pragma unroll
        for (int d = 1; d < 16; d <<= 1) v = fmaxf(v, __shfl_xor(v, d, 32));
        if (lrow == 0) {
          atomicMax(&lmax[mt2 * 32 + t * 16 + i + (hih ? 8 : 0)], encode_f32(v));  // ds_max
        }
      }
    }
    __syncthreads();   // protect aHi/aLo before next stage
  }

  // Merge workgroup-local maxima into the global per-row max.
#pragma unroll
  for (int i = 0; i < 8; ++i) {
    int r = tid * 8 + i;
    atomicMax(&gkey[r], lmax[r]);          // global_atomic_max_u32
  }
}

// ---------------------------------------------------------------------------
// Kernel 4: out[m] = 1 - max_sim[m]
// ---------------------------------------------------------------------------
__global__ void k_final(const unsigned* __restrict__ gkey, float* __restrict__ out) {
  int i = blockIdx.x * 256 + threadIdx.x;
  if (i < B_ROWS) out[i] = 1.0f - decode_f32(gkey[i]);
}

// ---------------------------------------------------------------------------
// Launch
// ---------------------------------------------------------------------------
extern "C" void kernel_launch(void* const* d_in, const int* in_sizes, int n_in,
                              void* d_out, int out_size, void* d_ws, size_t ws_size,
                              hipStream_t stream) {
  const float* x   = (const float*)d_in[0];   // [2048, 2048]
  const float* R   = (const float*)d_in[1];   // [2048, 256]
  const float* buf = (const float*)d_in[2];   // [131072, 256]
  float* out = (float*)d_out;                 // [2048]

  char* ws = (char*)d_ws;                     // needs ~4.01 MB
  __bf16*   Rth  = (__bf16*)(ws);                       // 1 MB
  __bf16*   Rtl  = (__bf16*)(ws + (size_t)(1u << 20));  // 1 MB
  __bf16*   sxh  = (__bf16*)(ws + (size_t)(2u << 20));  // 1 MB
  __bf16*   sxl  = (__bf16*)(ws + (size_t)(3u << 20));  // 1 MB
  unsigned* gkey = (unsigned*)(ws + (size_t)(4u << 20)); // 8 KB

  k_init      <<<(B_ROWS + 255) / 256, 256, 0, stream>>>(gkey);
  k_transposeR<<<(DIM * SK) / 256,     256, 0, stream>>>(R, Rth, Rtl);
  k_sketch    <<<B_ROWS / 16,          256, 0, stream>>>(x, Rth, Rtl, sxh, sxl);
  k_simmax    <<<NBUF / 128,           256, 0, stream>>>(sxh, sxl, buf, gkey);
  k_final     <<<(B_ROWS + 255) / 256, 256, 0, stream>>>(gkey, out);
}
